// RATIO_LVAE_shGLM_16097537425994
// MI455X (gfx1250) — matrix operand order repaired and checked by reading, not compiled, not run on hardware
//
#include <hip/hip_runtime.h>
#include <hip/hip_bf16.h>
#include <math.h>

// ---------------- Problem constants ----------------
#define T_DATA 20000
#define E_NO   2000
#define I_NO   500
#define SUB_NO 16
#define SYN_B  3
#define HID    256
#define T_SYN  200
#define T_ENC  80
#define W_IN   159          // 2*T_ENC - 1
#define W_PAD  160          // padded to multiple of 32
#define TM     128          // time rows per block in the MLP kernel

typedef __bf16 bf16_t;
typedef __attribute__((ext_vector_type(16))) __bf16 v16bf;
typedef __attribute__((ext_vector_type(8)))  float  v8f;
typedef __attribute__((ext_vector_type(4)))  float  v4f;   // native vector for NT loads

// =====================================================================
// Prep kernel 1: synaptic kernels ker_e/ker_i [16][200]
// =====================================================================
__global__ void ker_build_kernel(const float* __restrict__ K_syn,      // [16][3][3]
                                 const float* __restrict__ tau_syn,    // [3][3]
                                 const float* __restrict__ delta_syn,  // [16][2]
                                 float* __restrict__ ker_e,            // [16][200]
                                 float* __restrict__ ker_i) {
    int idx = blockIdx.x * blockDim.x + threadIdx.x;
    if (idx >= SUB_NO * T_SYN) return;
    int s = idx / T_SYN;
    int k = idx - s * T_SYN;
    float de = __expf(delta_syn[s * 2 + 0]);
    float di = __expf(delta_syn[s * 2 + 1]);
    float te = fmaxf((float)k - de, 0.0f);
    float ti = fmaxf((float)k - di, 0.0f);
    float se = 0.0f, si = 0.0f;
#pragma unroll
    for (int b = 0; b < SYN_B; ++b) {
        float taue = __expf(tau_syn[b * 3 + 0]);
        float taui = __expf(tau_syn[b * 3 + 1]);
        float tte = te / taue;
        float tti = ti / taui;
        se += tte * __expf(-tte) * K_syn[s * 9 + b * 3 + 0];
        si += tti * __expf(-tti) * K_syn[s * 9 + b * 3 + 1];
    }
    ker_e[idx] = se;
    ker_i[idx] = si;
}

// =====================================================================
// Prep kernel 2: one-hot -> subunit index
// =====================================================================
__global__ void assign_kernel(const float* __restrict__ Ce,   // [16][E_NO]
                              const float* __restrict__ Ci,   // [16][I_NO]
                              int* __restrict__ assign_e,
                              int* __restrict__ assign_i) {
    int idx = blockIdx.x * blockDim.x + threadIdx.x;
    if (idx < E_NO) {
        int a = 0;
#pragma unroll
        for (int s = 0; s < SUB_NO; ++s)
            if (Ce[s * E_NO + idx] > 0.5f) a = s;
        assign_e[idx] = a;
    } else if (idx < E_NO + I_NO) {
        int e = idx - E_NO;
        int a = 0;
#pragma unroll
        for (int s = 0; s < SUB_NO; ++s)
            if (Ci[s * I_NO + e] > 0.5f) a = s;
        assign_i[e] = a;
    }
}

// =====================================================================
// Prep kernel 3: f32 -> bf16 weight conversion with zero padding
// =====================================================================
__global__ void cvt_bf16_kernel(const float* __restrict__ src, bf16_t* __restrict__ dst,
                                int srcR, int srcC, int dstC, int total) {
    int idx = blockIdx.x * blockDim.x + threadIdx.x;
    if (idx >= total) return;
    int r = idx / dstC;
    int c = idx - r * dstC;
    float v = (r < srcR && c < srcC) ? src[r * srcC + c] : 0.0f;
    dst[idx] = (bf16_t)v;
}

// =====================================================================
// Spike aggregation: in_e[t][s], in_i[t][s] via LDS float atomics.
// Bandwidth-critical stage (~200MB): b128 non-temporal streaming loads
// (TH=NT keeps the once-read spike matrix from evicting L2-resident
// weights/activations); assignment lookups only on the ~2% nonzeros.
// =====================================================================
__global__ __launch_bounds__(256) void spike_agg_kernel(const v4f* __restrict__ S_e4,
                                                        const v4f* __restrict__ S_i4,
                                                        const int* __restrict__ ae,
                                                        const int* __restrict__ ai,
                                                        float* __restrict__ in_e,
                                                        float* __restrict__ in_i) {
    __shared__ float bins[2 * SUB_NO];
    size_t t = blockIdx.x;
    int tid = threadIdx.x;
    if (tid < 2 * SUB_NO) bins[tid] = 0.0f;
    __syncthreads();
    for (int q = tid; q < E_NO / 4; q += 256) {
        v4f v = __builtin_nontemporal_load(&S_e4[t * (E_NO / 4) + q]);
        int e = q * 4;
#pragma unroll
        for (int c = 0; c < 4; ++c)
            if (v[c] != 0.0f) atomicAdd(&bins[ae[e + c]], v[c]);   // ds_add_f32
    }
    for (int q = tid; q < I_NO / 4; q += 256) {
        v4f v = __builtin_nontemporal_load(&S_i4[t * (I_NO / 4) + q]);
        int i = q * 4;
#pragma unroll
        for (int c = 0; c < 4; ++c)
            if (v[c] != 0.0f) atomicAdd(&bins[SUB_NO + ai[i + c]], v[c]);
    }
    __syncthreads();
    if (tid < SUB_NO)
        in_e[t * SUB_NO + tid] = bins[tid];
    else if (tid < 2 * SUB_NO)
        in_i[t * SUB_NO + (tid - SUB_NO)] = bins[tid];
}

// =====================================================================
// Causal conv: S_conv[t][s] = sum_k in_e[t-k][s]*ker_e[s][k] + (i term)
// =====================================================================
__global__ __launch_bounds__(256) void causal_conv_kernel(const float* __restrict__ in_e,
                                                          const float* __restrict__ in_i,
                                                          const float* __restrict__ ker_e,
                                                          const float* __restrict__ ker_i,
                                                          float* __restrict__ S_conv) {
    int idx = blockIdx.x * blockDim.x + threadIdx.x;
    if (idx >= T_DATA * SUB_NO) return;
    int t = idx >> 4;
    int s = idx & 15;
    int kmax = t < (T_SYN - 1) ? t : (T_SYN - 1);
    float acc = 0.0f;
#pragma unroll 4
    for (int k = 0; k <= kmax; ++k) {
        acc = fmaf(in_e[(t - k) * SUB_NO + s], ker_e[s * T_SYN + k], acc);
        acc = fmaf(in_i[(t - k) * SUB_NO + s], ker_i[s * T_SYN + k], acc);
    }
    S_conv[idx] = acc;
}

// =====================================================================
// WMMA fragment loaders (contiguous k = 16*half + s layouts, wave32)
// =====================================================================
// A (16x32 bf16, M striped over lanes): lane l, m=l&15.
//   half=l>>4: elems 0..15 <- k = kb + 16*half + (0..15)
__device__ __forceinline__ v16bf load_a_frag(const bf16_t* __restrict__ S, int stride,
                                             int row0, int kb, int lane) {
    int half = lane >> 4;
    int m = row0 + (lane & 15);
    const bf16_t* p = S + m * stride + kb + 16 * half;
    v16bf a;
#pragma unroll
    for (int i = 0; i < 8; ++i) a[i] = p[i];
    const bf16_t* q = p + 8;
#pragma unroll
    for (int i = 0; i < 8; ++i) a[8 + i] = q[i];
    return a;
}

// B (32x16 bf16, N striped over lanes): lane l, n=ncol0+(l&15).
//   elems 0..15 <- k = kb + 16*half + (0..15)   (contiguous 32B)
// W is row-major [N_out][K], so B[k][n] = W[n*K + k].
__device__ __forceinline__ v16bf load_b_frag(const bf16_t* __restrict__ W, int K,
                                             int ncol0, int kb, int lane) {
    int half = lane >> 4;
    int n = ncol0 + (lane & 15);
    const bf16_t* p = W + n * K + kb + 16 * half;
    v16bf b;
#pragma unroll
    for (int i = 0; i < 16; ++i) b[i] = p[i];
    return b;
}

// Branch-free PReLU: max(x,0) + a*min(x,0)  (v_max/v_min/v_fmac)
__device__ __forceinline__ float prelu(float x, float a) {
    return fmaxf(x, 0.0f) + a * fminf(x, 0.0f);
}

// One dense layer (N=256): this wave's 16 rows x all 256 cols.
// sIn stride == K (160 for win, 256 for hidden). Fuses bias + PReLU,
// writes bf16 activations to sOut (stride 256).
__device__ __forceinline__ void dense_layer(const bf16_t* __restrict__ sIn, int K,
                                            const bf16_t* __restrict__ W,
                                            const float* __restrict__ bias, float alpha,
                                            bf16_t* __restrict__ sOut,
                                            int rowBase, int lane) {
    int half = lane >> 4;
    int l16 = lane & 15;
    for (int nt = 0; nt < HID / 16; ++nt) {
        int ncol = nt * 16;
        // head-start the next B panel through L2 -> WGP$ (global_prefetch_b8)
        if (nt + 1 < HID / 16)
            __builtin_prefetch(W + (ncol + 16 + l16) * K, 0, 1);
        v8f acc;
#pragma unroll
        for (int i = 0; i < 8; ++i) acc[i] = 0.0f;
        for (int kb = 0; kb < K; kb += 32) {
            v16bf a = load_a_frag(sIn, K, rowBase, kb, lane);
            v16bf b = load_b_frag(W, K, ncol, kb, lane);
            acc = __builtin_amdgcn_wmma_f32_16x16x32_bf16(
                false, a, false, b, (short)0, acc, false, false);
        }
        // D layout: VGPR r -> M = r + 8*half, N = lane&15
#pragma unroll
        for (int r = 0; r < 8; ++r) {
            int mm = rowBase + r + 8 * half;
            int nn = ncol + l16;
            float v = prelu(acc[r] + bias[nn], alpha);
            sOut[mm * HID + nn] = (bf16_t)v;
        }
    }
}

// =====================================================================
// Fused sliding-window MLP. Block = 256 threads = 8 waves; each wave
// owns 16 time rows (wave-private -> barrier-free layer loops). Window
// tile built from V_in in LDS; activations ping-pong LDS bf16 (performs
// the D->A relayout); layer4 fuses b4 + S_conv[:,1:] + enc_bias +
// sigmoid and writes the [T,15] output.
// =====================================================================
__global__ __launch_bounds__(256) void mlp_wmma_kernel(
        const float* __restrict__ V_in,
        const bf16_t* __restrict__ W1b, const float* __restrict__ b1, const float* __restrict__ pa1,
        const bf16_t* __restrict__ W2b, const float* __restrict__ b2, const float* __restrict__ pa2,
        const bf16_t* __restrict__ W3b, const float* __restrict__ b3, const float* __restrict__ pa3,
        const bf16_t* __restrict__ W4b, const float* __restrict__ b4,
        const float* __restrict__ S_conv, const float* __restrict__ enc_bias,
        float* __restrict__ out) {
    extern __shared__ char smem_raw[];
    bf16_t* sWin = (bf16_t*)smem_raw;           // [TM][W_PAD]
    bf16_t* sA   = sWin + TM * W_PAD;           // [TM][HID]
    bf16_t* sB   = sA + TM * HID;               // [TM][HID]

    int wave = threadIdx.x >> 5;
    int lane = threadIdx.x & 31;
    int half = lane >> 4;
    int l16  = lane & 15;
    int rowBase = wave * 16;
    int tBase = blockIdx.x * TM;

    // ---- build window tile: win[t][j] = V_in[t + j - (T_ENC-1)] or 0 ----
    for (int idx = lane; idx < 16 * W_PAD; idx += 32) {
        int r = idx / W_PAD;
        int j = idx - r * W_PAD;
        int t = tBase + rowBase + r;
        int vi = t + j - (T_ENC - 1);
        float v = (j < W_IN && t < T_DATA && vi >= 0 && vi < T_DATA) ? V_in[vi] : 0.0f;
        sWin[(rowBase + r) * W_PAD + j] = (bf16_t)v;
    }
    __syncthreads();

    float a1 = pa1[0], a2 = pa2[0], a3 = pa3[0];

    dense_layer(sWin, W_PAD, W1b, b1, a1, sA, rowBase, lane);   // [*,160]x[160,256]
    __syncthreads();
    dense_layer(sA, HID, W2b, b2, a2, sB, rowBase, lane);       // [*,256]x[256,256]
    __syncthreads();
    dense_layer(sB, HID, W3b, b3, a3, sA, rowBase, lane);       // [*,256]x[256,256]
    __syncthreads();

    // ---- layer 4: single N=16 tile (15 valid cols) + fused epilogue ----
    v8f acc;
#pragma unroll
    for (int i = 0; i < 8; ++i) acc[i] = 0.0f;
    for (int kb = 0; kb < HID; kb += 32) {
        v16bf a = load_a_frag(sA, HID, rowBase, kb, lane);
        v16bf b = load_b_frag(W4b, HID, 0, kb, lane);
        acc = __builtin_amdgcn_wmma_f32_16x16x32_bf16(
            false, a, false, b, (short)0, acc, false, false);
    }
#pragma unroll
    for (int r = 0; r < 8; ++r) {
        int mm = rowBase + r + 8 * half;
        int t = tBase + mm;
        int nn = l16;
        if (nn < (SUB_NO - 1) && t < T_DATA) {
            float v = acc[r] + b4[nn] + S_conv[t * SUB_NO + nn + 1] + enc_bias[nn];
            out[t * (SUB_NO - 1) + nn] = 1.0f / (1.0f + __expf(-v));
        }
    }
}

// =====================================================================
// Host-side launch
// =====================================================================
extern "C" void kernel_launch(void* const* d_in, const int* in_sizes, int n_in,
                              void* d_out, int out_size, void* d_ws, size_t ws_size,
                              hipStream_t stream) {
    const float* V_in      = (const float*)d_in[0];
    const float* S_e       = (const float*)d_in[1];
    const float* S_i       = (const float*)d_in[2];
    const float* C_syn_e   = (const float*)d_in[3];
    const float* C_syn_i   = (const float*)d_in[4];
    const float* K_syn     = (const float*)d_in[5];
    const float* tau_syn   = (const float*)d_in[6];
    const float* delta_syn = (const float*)d_in[7];
    const float* enc_bias  = (const float*)d_in[8];
    const float* W1 = (const float*)d_in[9];
    const float* b1 = (const float*)d_in[10];
    const float* a1 = (const float*)d_in[11];
    const float* W2 = (const float*)d_in[12];
    const float* b2 = (const float*)d_in[13];
    const float* a2 = (const float*)d_in[14];
    const float* W3 = (const float*)d_in[15];
    const float* b3 = (const float*)d_in[16];
    const float* a3 = (const float*)d_in[17];
    const float* W4 = (const float*)d_in[18];
    const float* b4 = (const float*)d_in[19];
    float* out = (float*)d_out;

    // ---- workspace carve-out (256B aligned) ----
    char* ws = (char*)d_ws;
    size_t off = 0;
    auto carve = [&](size_t bytes) -> void* {
        void* p = ws + off;
        off = (off + bytes + 255) & ~(size_t)255;
        return p;
    };
    float* ker_e    = (float*)carve(SUB_NO * T_SYN * sizeof(float));
    float* ker_i    = (float*)carve(SUB_NO * T_SYN * sizeof(float));
    int*   assign_e = (int*)carve(E_NO * sizeof(int));
    int*   assign_i = (int*)carve(I_NO * sizeof(int));
    float* in_e     = (float*)carve((size_t)T_DATA * SUB_NO * sizeof(float));
    float* in_i     = (float*)carve((size_t)T_DATA * SUB_NO * sizeof(float));
    float* S_conv   = (float*)carve((size_t)T_DATA * SUB_NO * sizeof(float));
    bf16_t* W1b     = (bf16_t*)carve((size_t)HID * W_PAD * sizeof(bf16_t));
    bf16_t* W2b     = (bf16_t*)carve((size_t)HID * HID * sizeof(bf16_t));
    bf16_t* W3b     = (bf16_t*)carve((size_t)HID * HID * sizeof(bf16_t));
    bf16_t* W4b     = (bf16_t*)carve((size_t)16 * HID * sizeof(bf16_t));

    // ---- prep ----
    ker_build_kernel<<<(SUB_NO * T_SYN + 255) / 256, 256, 0, stream>>>(
        K_syn, tau_syn, delta_syn, ker_e, ker_i);
    assign_kernel<<<(E_NO + I_NO + 255) / 256, 256, 0, stream>>>(
        C_syn_e, C_syn_i, assign_e, assign_i);
    cvt_bf16_kernel<<<(HID * W_PAD + 255) / 256, 256, 0, stream>>>(
        W1, W1b, HID, W_IN, W_PAD, HID * W_PAD);
    cvt_bf16_kernel<<<(HID * HID + 255) / 256, 256, 0, stream>>>(
        W2, W2b, HID, HID, HID, HID * HID);
    cvt_bf16_kernel<<<(HID * HID + 255) / 256, 256, 0, stream>>>(
        W3, W3b, HID, HID, HID, HID * HID);
    cvt_bf16_kernel<<<(16 * HID + 255) / 256, 256, 0, stream>>>(
        W4, W4b, SUB_NO - 1, HID, HID, 16 * HID);

    // ---- bandwidth stage: spike scatter (b128 NT streaming) ----
    spike_agg_kernel<<<T_DATA, 256, 0, stream>>>(
        (const v4f*)S_e, (const v4f*)S_i, assign_e, assign_i, in_e, in_i);

    // ---- causal convolution ----
    causal_conv_kernel<<<(T_DATA * SUB_NO + 255) / 256, 256, 0, stream>>>(
        in_e, in_i, ker_e, ker_i, S_conv);

    // ---- fused WMMA MLP ----
    size_t smem = (size_t)(TM * W_PAD + 2 * TM * HID) * sizeof(bf16_t);  // 168 KB
    int nblocks = (T_DATA + TM - 1) / TM;  // 157
    mlp_wmma_kernel<<<nblocks, 256, smem, stream>>>(
        V_in,
        W1b, b1, a1, W2b, b2, a2, W3b, b3, a3, W4b, b4,
        S_conv, enc_bias, out);
}
